// POLY2_73521250173556
// MI455X (gfx1250) — matrix operand order repaired and checked by reading, not compiled, not run on hardware
//
#include <hip/hip_runtime.h>
#include <stdint.h>

#if __has_builtin(__builtin_amdgcn_global_load_async_to_lds_b128) && \
    __has_builtin(__builtin_amdgcn_s_wait_asynccnt)
#define POLY2_ASYNC 1
#else
#define POLY2_ASYNC 0
#endif

typedef __attribute__((ext_vector_type(16))) __bf16 v16bf;
typedef __attribute__((ext_vector_type(8)))  float  v8f;
typedef __attribute__((ext_vector_type(4)))  int    v4i_t;

static constexpr int Bsz = 16384;
static constexpr int Nn  = 2048;
static constexpr int BM  = 64;    // rows per workgroup
static constexpr int BN  = 128;   // j-tile width
static constexpr int BK  = 32;    // k step (one bf16 WMMA K)
static constexpr int LDA = 20;    // dword stride of sA row (16 data + 4 pad, 16B-aligned)
static constexpr int LDB = 20;    // dword stride of sB j-row (16 data + 4 pad, 16B-aligned)

union FragBF {
    v16bf    v;
    uint32_t u[8];
    uint4    u4[2];
};

// f32 pair -> packed bf16 dword (lo = first). Round-half-up (<=0.5ulp) + one v_perm_b32.
__device__ __forceinline__ uint32_t pk_bf16(float lo, float hi) {
    uint32_t a = __float_as_uint(lo) + 0x8000u;
    uint32_t b = __float_as_uint(hi) + 0x8000u;
    return __builtin_amdgcn_perm(b, a, 0x07060302u);   // {b[31:16], a[31:16]}
}

// 16-byte global -> LDS copy; async (ASYNCcnt) on CDNA5, reg round-trip otherwise.
__device__ __forceinline__ void g2l16(const uint32_t* g, uint32_t* l) {
#if POLY2_ASYNC
    v4i_t* gv = (v4i_t*)(void*)const_cast<uint32_t*>(g);
    v4i_t* lv = (v4i_t*)(void*)l;
    __builtin_amdgcn_global_load_async_to_lds_b128(
        (__attribute__((address_space(1))) v4i_t*)gv,
        (__attribute__((address_space(3))) v4i_t*)lv, 0, 0);
#else
    *(uint4*)l = *(const uint4*)g;
#endif
}

__device__ __forceinline__ void async_publish_wait() {
#if POLY2_ASYNC
    __builtin_amdgcn_s_wait_asynccnt(0);
#endif
}

// ---- prep: pack w2 into bf16-pair dwords, layout [kb][j][kp]:
// dword o = (kb*Nn + j)*16 + kp  holds {K = kb*32+2kp (lo), +1 (hi)} at column j.
// Reads coalesced over j; scattered dword writes absorbed by L2 (one-time, 8 MiB).
__global__ __launch_bounds__(256) void w2_pack_kernel(
    const float* __restrict__ w2, uint32_t* __restrict__ w2p)
{
    const int idx = blockIdx.x * 256 + threadIdx.x;   // 0 .. 2M-1
    const int j  = idx & (Nn - 1);
    const int kp = (idx >> 11) & 15;
    const int kb = idx >> 15;
    const float* s = w2 + (size_t)(kb * 32 + 2 * kp) * Nn + j;
    w2p[((size_t)kb * Nn + j) * 16 + kp] = pk_bf16(s[0], s[Nn]);
}

// ---- prep: pack x row-major f32 -> bf16 pairs (row-major, [B][N/2] dwords) ----
__global__ __launch_bounds__(256) void x_pack_kernel(
    const float* __restrict__ x, uint32_t* __restrict__ xp)
{
    const size_t idx = (size_t)blockIdx.x * 256 + threadIdx.x;  // 0 .. 16M-1
    const float2 v = ((const float2*)x)[idx];
    xp[idx] = pk_bf16(v.x, v.y);
}

// PRE: 0 = stage from f32 (convert in-kernel), 1 = w2 pre-packed, 2 = w2 + x pre-packed
template <int PRE>
__global__ __launch_bounds__(256) void poly2_gemm_kernel(
    const float*    __restrict__ xg,   // [B, N] f32 (also epilogue source)
    const float*    __restrict__ w0,
    const float*    __restrict__ w1,
    const float*    __restrict__ w2f,  // f32 fallback staging source
    const uint32_t* __restrict__ w2p,  // packed bf16 pairs [kb][j][kp]
    const uint32_t* __restrict__ xp,   // packed bf16 pairs [row][kpair]
    float*          __restrict__ out)
{
    __shared__ uint32_t sA[2][BM * LDA];   // x tile:  [row][kpair]
    __shared__ uint32_t sB[2][BN * LDB];   // w2 tile: [col j][kpair]
    __shared__ float rowsum[BM];

    const int tid  = threadIdx.x;
    const int lane = tid & 31;
    const int wave = tid >> 5;
    const int half = lane >> 4;
    const int l15  = lane & 15;
    const int rw   = wave & 1;    // wave row block (2 x 32 rows)
    const int cw   = wave >> 1;   // wave col block (4 x 32 cols)
    const int gRow0 = blockIdx.x * BM;

    if (tid < BM) rowsum[tid] = 0.0f;

    // A staging: thread -> (row, 16B chunk)
    const int a_row = tid >> 2;
    const int a_c   = (tid & 3) * 8;                       // f32 col (PRE<2)
    const size_t a_srcf = (size_t)(gRow0 + a_row) * Nn + a_c;
    const uint32_t* a_srcp = xp + (size_t)(gRow0 + a_row) * (Nn / 2) + (tid & 3) * 4;
    const int a_dst = a_row * LDA + (tid & 3) * 4;         // 16B-aligned
    // B staging (PRE>=1): thread -> (j row, 16B chunk); covers j and j+64
    const int bj = tid >> 2;             // 0..63
    const int bq = (tid & 3) * 4;        // dword quarter within a j-row
    // B staging (PRE==0): thread -> (k pair b_kp, 8 consecutive j)
    const int b_kp = tid >> 4;
    const int b_tj = (tid & 15) * 8;

    float acc[2][8];
    #pragma unroll
    for (int i = 0; i < 2; ++i)
        #pragma unroll
        for (int r = 0; r < 8; ++r) acc[i][r] = 0.0f;

    for (int j0 = 0; j0 < Nn; j0 += BN) {
        v8f c00 = {}, c01 = {}, c10 = {}, c11 = {};

        // ---- pre-stage k0 = 0 into buffer 0 ----
        __syncthreads();   // WAR vs previous j-tile's last fragment reads
        {
            if (PRE == 2) {
                g2l16(a_srcp, &sA[0][a_dst]);
            } else {
                const float4* s = (const float4*)(xg + a_srcf);
                float4 f0 = s[0], f1 = s[1];
                uint4 t;
                t.x = pk_bf16(f0.x, f0.y); t.y = pk_bf16(f0.z, f0.w);
                t.z = pk_bf16(f1.x, f1.y); t.w = pk_bf16(f1.z, f1.w);
                *(uint4*)&sA[0][a_dst] = t;
            }
            if (PRE >= 1) {
                const uint32_t* bs = w2p + ((size_t)(j0 + bj)) * 16 + bq;
                g2l16(bs,        &sB[0][bj * LDB + bq]);
                g2l16(bs + 1024, &sB[0][(bj + 64) * LDB + bq]);
            } else {
                const float* r0 = w2f + (size_t)(2 * b_kp) * Nn + j0 + b_tj;
                const float4* q0 = (const float4*)r0;
                const float4* q1 = (const float4*)(r0 + Nn);
                float4 g0 = q0[0], g1 = q0[1], h0 = q1[0], h1 = q1[1];
                uint32_t* bd = &sB[0][b_tj * LDB + b_kp];
                bd[0 * LDB] = pk_bf16(g0.x, h0.x); bd[1 * LDB] = pk_bf16(g0.y, h0.y);
                bd[2 * LDB] = pk_bf16(g0.z, h0.z); bd[3 * LDB] = pk_bf16(g0.w, h0.w);
                bd[4 * LDB] = pk_bf16(g1.x, h1.x); bd[5 * LDB] = pk_bf16(g1.y, h1.y);
                bd[6 * LDB] = pk_bf16(g1.z, h1.z); bd[7 * LDB] = pk_bf16(g1.w, h1.w);
            }
        }

        int p = 0;
        #pragma unroll 2
        for (int k0 = 0; k0 < Nn; k0 += BK) {
            async_publish_wait();   // our async copies into buffer p are in LDS
            __syncthreads();        // all waves' stage of buffer p visible
            const bool more = (k0 + BK) < Nn;
            const int  kn   = k0 + BK;

            // ---- issue next-stage copies / loads early (into buffer p^1) ----
            float4 nA0 = {}, nA1 = {};
            float4 nG0 = {}, nG1 = {}, nH0 = {}, nH1 = {};
            if (more) {
                if (PRE == 2) {
                    g2l16(a_srcp + (kn >> 1), &sA[p ^ 1][a_dst]);
                } else {
                    const float4* s = (const float4*)(xg + a_srcf + kn);
                    nA0 = s[0]; nA1 = s[1];
                }
                if (PRE >= 1) {
                    const uint32_t* bs =
                        w2p + ((size_t)(kn >> 5) * Nn + j0 + bj) * 16 + bq;
                    g2l16(bs,        &sB[p ^ 1][bj * LDB + bq]);
                    g2l16(bs + 1024, &sB[p ^ 1][(bj + 64) * LDB + bq]);
                } else {
                    const float* r0 = w2f + (size_t)(kn + 2 * b_kp) * Nn + j0 + b_tj;
                    const float4* q0 = (const float4*)r0;
                    const float4* q1 = (const float4*)(r0 + Nn);
                    nG0 = q0[0]; nG1 = q0[1]; nH0 = q1[0]; nH1 = q1[1];
                }
            }

            // ---- fragments from buffer p (all aligned b128 DS loads) ----
            FragBF a0, a1, b0, b1;
            {
                const int r0i = (rw * 32 + l15) * LDA + 4 * half;
                a0.u4[0] = *(const uint4*)&sA[p][r0i];
                a0.u4[1] = *(const uint4*)&sA[p][r0i + 8];
                const int r1i = r0i + 16 * LDA;
                a1.u4[0] = *(const uint4*)&sA[p][r1i];
                a1.u4[1] = *(const uint4*)&sA[p][r1i + 8];
            }
            {
                const int nb0 = (cw * 32 + l15) * LDB + 4 * half;
                b0.u4[0] = *(const uint4*)&sB[p][nb0];
                b0.u4[1] = *(const uint4*)&sB[p][nb0 + 8];
                const int nb1 = nb0 + 16 * LDB;
                b1.u4[0] = *(const uint4*)&sB[p][nb1];
                b1.u4[1] = *(const uint4*)&sB[p][nb1 + 8];
            }

            c00 = __builtin_amdgcn_wmma_f32_16x16x32_bf16(false, a0.v, false, b0.v,
                                                          (short)0, c00, false, false);
            c01 = __builtin_amdgcn_wmma_f32_16x16x32_bf16(false, a0.v, false, b1.v,
                                                          (short)0, c01, false, false);
            c10 = __builtin_amdgcn_wmma_f32_16x16x32_bf16(false, a1.v, false, b0.v,
                                                          (short)0, c10, false, false);
            c11 = __builtin_amdgcn_wmma_f32_16x16x32_bf16(false, a1.v, false, b1.v,
                                                          (short)0, c11, false, false);

            // ---- deferred stores for register-pipelined paths ----
            if (more) {
                if (PRE < 2) {
                    uint4 t;
                    t.x = pk_bf16(nA0.x, nA0.y); t.y = pk_bf16(nA0.z, nA0.w);
                    t.z = pk_bf16(nA1.x, nA1.y); t.w = pk_bf16(nA1.z, nA1.w);
                    *(uint4*)&sA[p ^ 1][a_dst] = t;
                }
                if (PRE == 0) {
                    uint32_t* bd = &sB[p ^ 1][b_tj * LDB + b_kp];
                    bd[0 * LDB] = pk_bf16(nG0.x, nH0.x); bd[1 * LDB] = pk_bf16(nG0.y, nH0.y);
                    bd[2 * LDB] = pk_bf16(nG0.z, nH0.z); bd[3 * LDB] = pk_bf16(nG0.w, nH0.w);
                    bd[4 * LDB] = pk_bf16(nG1.x, nH1.x); bd[5 * LDB] = pk_bf16(nG1.y, nH1.y);
                    bd[6 * LDB] = pk_bf16(nG1.z, nH1.z); bd[7 * LDB] = pk_bf16(nG1.w, nH1.w);
                }
            }
            p ^= 1;
        } // k0

        // ---- fused epilogue: acc += (y + w1) * x  (C layout addressing) ----
        {
            const int n0 = j0 + cw * 32 + l15;
            const float w1v0 = w1[n0];
            const float w1v1 = w1[n0 + 16];

            const float* px0 = xg + (size_t)(gRow0 + rw * 32 + 8 * half) * Nn + n0;
            #pragma unroll
            for (int r = 0; r < 8; ++r) {
                float xv0 = px0[(size_t)r * Nn];
                float xv1 = px0[(size_t)r * Nn + 16];
                acc[0][r] += (c00[r] + w1v0) * xv0 + (c01[r] + w1v1) * xv1;
            }
            const float* px1 = px0 + (size_t)16 * Nn;
            #pragma unroll
            for (int r = 0; r < 8; ++r) {
                float xv0 = px1[(size_t)r * Nn];
                float xv1 = px1[(size_t)r * Nn + 16];
                acc[1][r] += (c10[r] + w1v0) * xv0 + (c11[r] + w1v1) * xv1;
            }
        }
    } // j0

    // ---- reduce N-groups (16 lanes), then across waves via LDS atomics ----
    #pragma unroll
    for (int i = 0; i < 2; ++i) {
        #pragma unroll
        for (int r = 0; r < 8; ++r) {
            float v = acc[i][r];
            v += __shfl_xor(v, 1);
            v += __shfl_xor(v, 2);
            v += __shfl_xor(v, 4);
            v += __shfl_xor(v, 8);
            if (l15 == 0)
                atomicAdd(&rowsum[rw * 32 + 16 * i + 8 * half + r], v);
        }
    }
    __syncthreads();

    if (tid < BM) {
        float v = w0[0] + rowsum[tid];
        out[gRow0 + tid] = 1.0f / (1.0f + __expf(-v));
    }
}

extern "C" void kernel_launch(void* const* d_in, const int* in_sizes, int n_in,
                              void* d_out, int out_size, void* d_ws, size_t ws_size,
                              hipStream_t stream) {
    const float* x  = (const float*)d_in[0];
    const float* w0 = (const float*)d_in[1];
    const float* w1 = (const float*)d_in[2];
    const float* w2 = (const float*)d_in[3];
    float* out = (float*)d_out;
    (void)in_sizes; (void)n_in; (void)out_size;

    const size_t W2P_BYTES = (size_t)Nn * Nn * 2;   // 8 MiB packed w2
    const size_t XP_BYTES  = (size_t)Bsz * Nn * 2;  // 64 MiB packed x
    uint32_t* w2p = (uint32_t*)d_ws;
    uint32_t* xp  = (uint32_t*)((char*)d_ws + W2P_BYTES);

    dim3 grid(Bsz / BM);   // 256 workgroups
    dim3 block(256);       // 8 wave32 waves

    if (ws_size >= W2P_BYTES + XP_BYTES) {
        w2_pack_kernel<<<dim3((Nn * Nn / 2) / 256), block, 0, stream>>>(w2, w2p);
        x_pack_kernel<<<dim3((int)(((size_t)Bsz * Nn / 2) / 256)), block, 0, stream>>>(x, xp);
        poly2_gemm_kernel<2><<<grid, block, 0, stream>>>(x, w0, w1, w2, w2p, xp, out);
    } else if (ws_size >= W2P_BYTES) {
        w2_pack_kernel<<<dim3((Nn * Nn / 2) / 256), block, 0, stream>>>(w2, w2p);
        poly2_gemm_kernel<1><<<grid, block, 0, stream>>>(x, w0, w1, w2, w2p, nullptr, out);
    } else {
        poly2_gemm_kernel<0><<<grid, block, 0, stream>>>(x, w0, w1, w2, nullptr, nullptr, out);
    }
}
